// SRGNNLayer_56556129353759
// MI455X (gfx1250) — matrix operand order repaired and checked by reading, not compiled, not run on hardware
//
#include <hip/hip_runtime.h>
#include <hip/hip_bf16.h>

typedef __attribute__((ext_vector_type(16))) __bf16 v16bf;
typedef __attribute__((ext_vector_type(8)))  float  v8f;

#define NGRAPH 8192
#define NP     32
#define HDIM   128
#define NEDGE  524288

static __device__ __forceinline__ v8f wmma_bf16(v16bf a, v16bf b, v8f c) {
  return __builtin_amdgcn_wmma_f32_16x16x32_bf16(false, a, false, b, (short)0, c, false, false);
}

// A fragment 16x32 (MxK) from row-major bf16 buffer (LDS).
// lane<16: M=lane, K = k0+0..15; lane>=16: M=lane-16, K = k0+16..31
static __device__ __forceinline__ v16bf load_a_bf(const __bf16* base, int row0, int ld, int k0, int lane) {
  int h = lane >> 4;
  const __bf16* p = base + (row0 + (lane & 15)) * ld + k0 + 16 * h;
  v16bf a;
#pragma unroll
  for (int e = 0; e < 8; ++e) a[e] = p[e];
#pragma unroll
  for (int e = 0; e < 8; ++e) a[8 + e] = p[8 + e];
  return a;
}

// B fragment 32x16 (KxN) with B[k][n] = W[n][k], W row-major [N][ldk] bf16 (global).
// lanes 0-15 hold K=k0..k0+15, lanes 16-31 hold K=k0+16..k0+31 (N = lane&15).
static __device__ __forceinline__ v16bf load_b_w(const __bf16* W, int ldk, int n0, int k0, int lane) {
  int h = lane >> 4;
  const __bf16* p = W + (n0 + (lane & 15)) * ldk + k0 + 16 * h;
  v16bf b;
#pragma unroll
  for (int e = 0; e < 16; ++e) b[e] = p[e];
  return b;
}

// fast transcendentals on v_exp_f32 / v_rcp_f32 (correct saturation at +/-inf)
static __device__ __forceinline__ float fast_sigmoid(float x) {
  return __builtin_amdgcn_rcpf(1.0f + __builtin_amdgcn_exp2f(-1.442695041f * x));
}
static __device__ __forceinline__ float fast_tanh(float x) {
  return 1.0f - 2.0f * __builtin_amdgcn_rcpf(1.0f + __builtin_amdgcn_exp2f(2.885390082f * x));
}

// ---------------- edge bucketing ----------------

__global__ void zero_cursor_kernel(int* cursor) {
  cursor[blockIdx.x * 256 + threadIdx.x] = 0;
}

__global__ void count_edges_kernel(const int* __restrict__ src, int* __restrict__ cursor) {
  int e = blockIdx.x * 256 + threadIdx.x;
  atomicAdd(&cursor[src[e] >> 5], 1);
}

// single block, 256 threads: exclusive scan over 8192 counts
__global__ void scan_offsets_kernel(int* __restrict__ cursor, int* __restrict__ graph_off) {
  __shared__ int part[256];
  int t = threadIdx.x;
  int base = t * 32;
  int vals[32];
  int s = 0;
#pragma unroll
  for (int i = 0; i < 32; ++i) { vals[i] = cursor[base + i]; s += vals[i]; }
  part[t] = s;
  __syncthreads();
  for (int off = 1; off < 256; off <<= 1) {
    int v = part[t];
    int add = (t >= off) ? part[t - off] : 0;
    __syncthreads();
    part[t] = v + add;
    __syncthreads();
  }
  int run = (t == 0) ? 0 : part[t - 1];
#pragma unroll
  for (int i = 0; i < 32; ++i) {
    graph_off[base + i] = run;
    cursor[base + i] = run;
    run += vals[i];
  }
  if (t == 255) graph_off[NGRAPH] = run;
}

__global__ void scatter_edges_kernel(const int* __restrict__ src, const int* __restrict__ dst,
                                     int* __restrict__ cursor, int* __restrict__ edge_packed) {
  int e = blockIdx.x * 256 + threadIdx.x;
  int sv = src[e], dv = dst[e];
  int g = sv >> 5;
  int pos = atomicAdd(&cursor[g], 1);
  edge_packed[pos] = ((sv & 31) << 5) | (dv & 31);
}

__global__ void f32_to_bf16_kernel(const float* __restrict__ s, __bf16* __restrict__ d, int n) {
  int i = blockIdx.x * 256 + threadIdx.x;
  if (i < n) d[i] = (__bf16)s[i];
}

// ---------------- fused per-graph kernel ----------------
// LDS layout (58368 B total):
//   P [0, 8192)      : feat bf16 [32][128]  -> later hn bf16
//   Q [8192, 24576)  : feat_in/feat_out bf16 [32][128]x2 -> a bf16 [32][256] -> hn f32 [32][128]
//   R [24576, 57344) : a f32 accum [32][256] -> gates bf16 (rpre,zpre,ni,nh)[32][128] -> feat_u f32 [32][128]
//   S [57344, 58368) : feat_v[128] f32, deg_in[32], deg_out[32], alpha[32]

__global__ __launch_bounds__(256)
void srgnn_graph_kernel(const float* __restrict__ feat, const float* __restrict__ cnt,
                        const __bf16* __restrict__ W_in_b, const float* __restrict__ b_in,
                        const __bf16* __restrict__ W_og_b, const float* __restrict__ b_og,
                        const __bf16* __restrict__ W_ih_b, const float* __restrict__ b_ih,
                        const __bf16* __restrict__ W_hh_b, const float* __restrict__ b_hh,
                        const __bf16* __restrict__ W_u_b,  const float* __restrict__ W_v,
                        const float* __restrict__ b_v, const float* __restrict__ w_e,
                        const int* __restrict__ edge_packed, const int* __restrict__ graph_off,
                        float* __restrict__ out) {
  __shared__ __align__(16) unsigned char smem[58368];
  __bf16* s_featb = (__bf16*)(smem);                    // [32][128], later hn bf16
  __bf16* s_hnb   = s_featb;
  __bf16* s_finb  = (__bf16*)(smem + 8192);             // [32][128]
  __bf16* s_foutb = (__bf16*)(smem + 16384);            // [32][128]
  __bf16* s_abf   = (__bf16*)(smem + 8192);             // [32][256] (after scatter)
  float*  s_hn    = (float*)(smem + 8192);              // [32][128] (after GEMM2)
  float*  s_a     = (float*)(smem + 24576);             // [32][256] f32 accum
  __bf16* s_rpre  = (__bf16*)(smem + 24576);            // [32][128]
  __bf16* s_zpre  = (__bf16*)(smem + 24576 + 8192);
  __bf16* s_ni    = (__bf16*)(smem + 24576 + 16384);
  __bf16* s_nh    = (__bf16*)(smem + 24576 + 24576);
  float*  s_fu    = (float*)(smem + 24576);             // [32][128]
  float*  s_fv    = (float*)(smem + 57344);             // [128]
  float*  s_degin  = s_fv + 128;                        // [32]
  float*  s_degout = s_degin + 32;                      // [32]
  float*  s_alpha  = s_degout + 32;                     // [32]

  const int g = blockIdx.x;
  const int tid = threadIdx.x;
  const int lane = tid & 31;
  const int wave = tid >> 5;
  const int rowbase = g * NP;
  const v8f zv = {0.f, 0.f, 0.f, 0.f, 0.f, 0.f, 0.f, 0.f};

  // stage 0: load feat tile to bf16 LDS; zero accumulators
  for (int i = tid; i < NP * HDIM; i += 256)
    s_featb[i] = (__bf16)feat[rowbase * HDIM + i];
  for (int i = tid; i < NP * 256; i += 256)
    s_a[i] = 0.0f;
  if (tid < 32) { s_degin[tid] = 0.0f; s_degout[tid] = 0.0f; }
  __syncthreads();

  // stage 1: GEMM feat_in / feat_out  (M=32, N=128, K=128)
  {
    int sel = wave >> 2;       // 0: feat_in, 1: feat_out
    int ww = wave & 3;         // ntile pair
    const __bf16* W = sel ? W_og_b : W_in_b;
    const float* bias = sel ? b_og : b_in;
    __bf16* dstb = sel ? s_foutb : s_finb;
    v8f acc[2][2];
#pragma unroll
    for (int mt = 0; mt < 2; ++mt)
#pragma unroll
      for (int j = 0; j < 2; ++j) acc[mt][j] = zv;
#pragma unroll
    for (int ks = 0; ks < 4; ++ks) {
      int k0 = ks * 32;
      v16bf a0 = load_a_bf(s_featb, 0, HDIM, k0, lane);
      v16bf a1 = load_a_bf(s_featb, 16, HDIM, k0, lane);
#pragma unroll
      for (int j = 0; j < 2; ++j) {
        v16bf b = load_b_w(W, HDIM, (ww * 2 + j) * 16, k0, lane);
        acc[0][j] = wmma_bf16(a0, b, acc[0][j]);
        acc[1][j] = wmma_bf16(a1, b, acc[1][j]);
      }
    }
    int h = lane >> 4, nn = lane & 15;
#pragma unroll
    for (int mt = 0; mt < 2; ++mt)
#pragma unroll
      for (int j = 0; j < 2; ++j) {
        int n = (ww * 2 + j) * 16 + nn;
        float bv = bias[n];
#pragma unroll
        for (int r = 0; r < 8; ++r) {
          int m = mt * 16 + 8 * h + r;
          dstb[m * HDIM + n] = (__bf16)(acc[mt][j][r] + bv);
        }
      }
  }
  __syncthreads();

  // stage 2: intra-graph edge scatter entirely in LDS (ds_add_f32 atomics)
  {
    int e0 = graph_off[g];
    int ne = graph_off[g + 1] - e0;
    int col = tid & 127;
    int sub = tid >> 7;   // 0..1 : 2 edges per iteration
    for (int base = 0; base < ne; base += 2) {
      int idx = base + sub;
      if (idx < ne) {
        int pk = edge_packed[e0 + idx];
        int sl = (pk >> 5) & 31;
        int dl = pk & 31;
        atomicAdd(&s_a[dl * 256 + col], (float)s_finb[sl * HDIM + col]);
        atomicAdd(&s_a[sl * 256 + 128 + col], (float)s_foutb[dl * HDIM + col]);
        if (col == 0) {
          atomicAdd(&s_degin[dl], 1.0f);
          atomicAdd(&s_degout[sl], 1.0f);
        }
      }
    }
  }
  __syncthreads();

  // normalize a by degree and convert ONCE to bf16 into Q (fin/fout are dead).
  // i = tid + 256*k  =>  col = tid (wave-uniform branch), row = k
  for (int k = 0; k < NP; ++k) {
    int i = tid + (k << 8);
    float d = (tid < 128) ? s_degin[k] : s_degout[k];
    s_abf[i] = (__bf16)(s_a[i] * __builtin_amdgcn_rcpf(fmaxf(d, 1.0f)));
  }
  __syncthreads();

  // stage 3: GEMM gi = a @ W_ih^T (K=256, A from s_abf) and gh = feat @ W_hh^T (K=128); N=384
  {
    v8f gi[2][3], gh[2][3];
#pragma unroll
    for (int mt = 0; mt < 2; ++mt)
#pragma unroll
      for (int j = 0; j < 3; ++j) { gi[mt][j] = zv; gh[mt][j] = zv; }

#pragma unroll
    for (int ks = 0; ks < 8; ++ks) {
      int k0 = ks * 32;
      v16bf a0 = load_a_bf(s_abf, 0, 256, k0, lane);
      v16bf a1 = load_a_bf(s_abf, 16, 256, k0, lane);
#pragma unroll
      for (int j = 0; j < 3; ++j) {
        v16bf b = load_b_w(W_ih_b, 256, (wave * 3 + j) * 16, k0, lane);
        gi[0][j] = wmma_bf16(a0, b, gi[0][j]);
        gi[1][j] = wmma_bf16(a1, b, gi[1][j]);
      }
    }
#pragma unroll
    for (int ks = 0; ks < 4; ++ks) {
      int k0 = ks * 32;
      v16bf a0 = load_a_bf(s_featb, 0, HDIM, k0, lane);
      v16bf a1 = load_a_bf(s_featb, 16, HDIM, k0, lane);
#pragma unroll
      for (int j = 0; j < 3; ++j) {
        v16bf b = load_b_w(W_hh_b, HDIM, (wave * 3 + j) * 16, k0, lane);
        gh[0][j] = wmma_bf16(a0, b, gh[0][j]);
        gh[1][j] = wmma_bf16(a1, b, gh[1][j]);
      }
    }

    // gate stores go to R (a f32 accum: dead since the normalize barrier); no extra barrier needed
    int h = lane >> 4, nn = lane & 15;
#pragma unroll
    for (int j = 0; j < 3; ++j) {
      int n = (wave * 3 + j) * 16 + nn;
      int gate = n >> 7;          // uniform per tile (tiles never straddle gates)
      int c = n & 127;
      float bi = b_ih[n], bh = b_hh[n];
#pragma unroll
      for (int mt = 0; mt < 2; ++mt)
#pragma unroll
        for (int r = 0; r < 8; ++r) {
          int m = mt * 16 + 8 * h + r;
          float vi = gi[mt][j][r] + bi;
          float vh = gh[mt][j][r] + bh;
          if (gate == 0) s_rpre[m * HDIM + c] = (__bf16)(vi + vh);
          else if (gate == 1) s_zpre[m * HDIM + c] = (__bf16)(vi + vh);
          else { s_ni[m * HDIM + c] = (__bf16)vi; s_nh[m * HDIM + c] = (__bf16)vh; }
        }
    }
  }
  __syncthreads();

  // stage 4: GRU elementwise -> hn (fp32 to Q over a_bf, bf16 to P over feat)
  for (int i = tid; i < NP * HDIM; i += 256) {
    float r = fast_sigmoid((float)s_rpre[i]);
    float z = fast_sigmoid((float)s_zpre[i]);
    float ng = fast_tanh((float)s_ni[i] + r * (float)s_nh[i]);
    float f = (float)s_featb[i];
    float hnv = (1.0f - z) * ng + z * f;
    s_hn[i] = hnv;
    s_hnb[i] = (__bf16)hnv;   // same slot as s_featb[i]; read-before-write in-thread
  }
  __syncthreads();

  // stage 5: feat_v = hn[31] @ W_v^T + b_v   (small matvec, VALU)
  if (tid < HDIM) {
    const float* wr = W_v + tid * HDIM;
    float s = 0.0f;
#pragma unroll 4
    for (int k = 0; k < HDIM; ++k) s += s_hn[31 * HDIM + k] * wr[k];
    s_fv[tid] = s + b_v[tid];
  }
  __syncthreads();

  // stage 6: GEMM feat_u = hn @ W_u^T  (M=32, N=128, K=128) -> R region (gates dead)
  {
    int mt = wave >> 2;
    int wq = wave & 3;
    v8f acc[2];
    acc[0] = zv; acc[1] = zv;
#pragma unroll
    for (int ks = 0; ks < 4; ++ks) {
      int k0 = ks * 32;
      v16bf a = load_a_bf(s_hnb, mt * 16, HDIM, k0, lane);
#pragma unroll
      for (int j = 0; j < 2; ++j) {
        v16bf b = load_b_w(W_u_b, HDIM, (wq * 2 + j) * 16, k0, lane);
        acc[j] = wmma_bf16(a, b, acc[j]);
      }
    }
    int h = lane >> 4, nn = lane & 15;
#pragma unroll
    for (int j = 0; j < 2; ++j) {
      int n = (wq * 2 + j) * 16 + nn;
#pragma unroll
      for (int r = 0; r < 8; ++r) {
        int m = mt * 16 + 8 * h + r;
        s_fu[m * HDIM + n] = acc[j][r];
      }
    }
  }
  __syncthreads();

  // stage 7: e = sigmoid(feat_u + feat_v) @ w_e ; alpha = e * cnt
  if (tid < 32) {
    float s = 0.0f;
#pragma unroll 4
    for (int k = 0; k < HDIM; ++k) {
      float x = s_fu[tid * HDIM + k] + s_fv[k];
      s += w_e[k] * fast_sigmoid(x);
    }
    s_alpha[tid] = s * cnt[rowbase + tid];
  }
  __syncthreads();

  // stage 8: ct_g = sum_m hn[m]*alpha[m]; ct_l = hn[31]; write [ct_g | ct_l]
  if (tid < HDIM) {
    float acc = 0.0f;
#pragma unroll
    for (int m = 0; m < NP; ++m) acc += s_hn[m * HDIM + tid] * s_alpha[m];
    out[g * (2 * HDIM) + tid] = acc;
    out[g * (2 * HDIM) + HDIM + tid] = s_hn[31 * HDIM + tid];
  }
}

// ---------------- launch ----------------

extern "C" void kernel_launch(void* const* d_in, const int* in_sizes, int n_in,
                              void* d_out, int out_size, void* d_ws, size_t ws_size,
                              hipStream_t stream) {
  (void)in_sizes; (void)n_in; (void)out_size; (void)ws_size;
  const float* feat = (const float*)d_in[0];
  const float* cnt  = (const float*)d_in[1];
  const float* W_in = (const float*)d_in[2];
  const float* b_in = (const float*)d_in[3];
  const float* W_og = (const float*)d_in[4];
  const float* b_og = (const float*)d_in[5];
  const float* W_ih = (const float*)d_in[6];
  const float* b_ih = (const float*)d_in[7];
  const float* W_hh = (const float*)d_in[8];
  const float* b_hh = (const float*)d_in[9];
  const float* W_u  = (const float*)d_in[10];
  const float* W_v  = (const float*)d_in[11];
  const float* b_v  = (const float*)d_in[12];
  const float* w_e  = (const float*)d_in[13];
  const int* src = (const int*)d_in[14];
  const int* dst = (const int*)d_in[15];
  float* out = (float*)d_out;

  char* ws = (char*)d_ws;
  int* edge_packed = (int*)ws;                                   // NEDGE ints
  int* graph_off   = (int*)(ws + (size_t)NEDGE * 4);             // NGRAPH+1 ints
  int* cursor      = (int*)(ws + (size_t)NEDGE * 4 + 33024);     // NGRAPH ints
  __bf16* wb       = (__bf16*)(ws + (size_t)NEDGE * 4 + 33024 + 32768);
  __bf16* W_in_b = wb;              // 16384
  __bf16* W_og_b = wb + 16384;      // 16384
  __bf16* W_ih_b = wb + 32768;      // 98304
  __bf16* W_hh_b = wb + 131072;     // 49152
  __bf16* W_u_b  = wb + 180224;     // 16384

  // weight fp32 -> bf16 (L2-resident, read by every block)
  f32_to_bf16_kernel<<<64, 256, 0, stream>>>(W_in, W_in_b, 16384);
  f32_to_bf16_kernel<<<64, 256, 0, stream>>>(W_og, W_og_b, 16384);
  f32_to_bf16_kernel<<<384, 256, 0, stream>>>(W_ih, W_ih_b, 98304);
  f32_to_bf16_kernel<<<192, 256, 0, stream>>>(W_hh, W_hh_b, 49152);
  f32_to_bf16_kernel<<<64, 256, 0, stream>>>(W_u, W_u_b, 16384);

  // bucket edges by graph
  zero_cursor_kernel<<<NGRAPH / 256, 256, 0, stream>>>(cursor);
  count_edges_kernel<<<NEDGE / 256, 256, 0, stream>>>(src, cursor);
  scan_offsets_kernel<<<1, 256, 0, stream>>>(cursor, graph_off);
  scatter_edges_kernel<<<NEDGE / 256, 256, 0, stream>>>(src, dst, cursor, edge_packed);

  // fused per-graph SR-GNN layer
  srgnn_graph_kernel<<<NGRAPH, 256, 0, stream>>>(
      feat, cnt, W_in_b, b_in, W_og_b, b_og, W_ih_b, b_ih, W_hh_b, b_hh,
      W_u_b, W_v, b_v, w_e, edge_packed, graph_off, out);
}
